// PerceptionLayer_16793322127864
// MI455X (gfx1250) — compile-verified, matrix operand order checked by reading, not run on hardware
//
#include <hip/hip_runtime.h>
#include <hip/hip_bf16.h>
#include <math.h>

// ---------------------------------------------------------------------------
// Transformer layer (pre-norm causal linear attention + GELU FFN) for gfx1250.
// B=1, S=2048, D=1024, H=16, HD=64, FF=4096.  f16 WMMA, f32 accumulate.
// GEMM tiles staged global->LDS via the Tensor Data Mover (TENSORcnt).
// ---------------------------------------------------------------------------

typedef __attribute__((ext_vector_type(16))) _Float16 v16h;
typedef __attribute__((ext_vector_type(8)))  _Float16 v8h;
typedef __attribute__((ext_vector_type(8)))  float    v8f;
typedef __attribute__((ext_vector_type(4))) unsigned  u32x4;
typedef __attribute__((ext_vector_type(4)))  int      i32x4;
typedef __attribute__((ext_vector_type(8)))  int      i32x8;

#define SEQ  2048
#define DM   1024
#define NH   16
#define HD   64
#define FFD  4096

__device__ __forceinline__ v8f wmma16(v16h a, v16h b, v8f c) {
    // D = A(16x32 f16) * B(32x16 f16) + C(16x16 f32)
    return __builtin_amdgcn_wmma_f32_16x16x32_f16(
        /*neg_a=*/false, a, /*neg_b=*/false, b,
        /*c_mod=*/(short)0, c, /*reuse_a=*/false, /*reuse_b=*/false);
}

// Load one 16x32 A/B fragment row from LDS (row-major, 'ldt' f16 stride).
// lanes0-15: VGPR0-3 K=[0..8), VGPR4-7 K=[16..24); lanes16-31: +8.
__device__ __forceinline__ v16h ld_frag(const _Float16* base, int row, int k0,
                                        int ldt, int half) {
    const _Float16* p = base + row * ldt + k0 + half * 8;
    v8h lo = *(const v8h*)(p);
    v8h hi = *(const v8h*)(p + 16);
    v16h r;
#pragma unroll
    for (int i = 0; i < 8; ++i) { r[i] = lo[i]; r[i + 8] = hi[i]; }
    return r;
}

// ---------------------------------------------------------------------------
// TDM: DMA one (rows x 32 f16) 2D tile, row stride K f16, into LDS with
// +16B pad per 64B row (LDS stride 40 f16 == G_LDT).  Issued per-wave,
// tracked with TENSORcnt.
// ---------------------------------------------------------------------------
__device__ __forceinline__ void tdm_load_tile32(unsigned lds_addr,
                                                const void* gptr,
                                                unsigned rows,
                                                unsigned stride_f16) {
    unsigned long long ga = (unsigned long long)(uintptr_t)gptr;
    u32x4 g0;
    g0[0] = 1u;                                   // count=1, user descriptor
    g0[1] = lds_addr;                             // LDS byte address
    g0[2] = (unsigned)ga;                         // global_addr[31:0]
    g0[3] = (unsigned)((ga >> 32) & 0x01ffffffu)  // global_addr[56:32]
          | (2u << 30);                           // type = 2 (image)
    i32x8 g1;
    g1[0] = (1 << 16)                             // data_size = 2 bytes
          | (1 << 20)                             // pad_enable
          | (3 << 22)                             // pad_interval: 16 DWORDs
          | (3 << 25);                            // pad_amount:   4 DWORDs
    g1[1] = (int)((stride_f16 & 0xffffu) << 16);          // tensor_dim0[15:0]
    g1[2] = (int)((stride_f16 >> 16) & 0xffffu)           // tensor_dim0[31:16]
          | (int)(rows << 16);                            // tensor_dim1[15:0]
    g1[3] = (int)(32u << 16);                             // tile_dim0 = 32
    g1[4] = (int)rows;                                    // tile_dim1
    g1[5] = (int)stride_f16;                              // dim0_stride[31:0]
    g1[6] = 0;
    g1[7] = 0;
    i32x4 z4 = {};
#if __has_include(<hip/amd_detail/amd_gfx1250_TDM.h>)
    i32x8 z8 = {};
    __builtin_amdgcn_tensor_load_to_lds(g0, g1, z4, z4, z8, 0);
#else
    __builtin_amdgcn_tensor_load_to_lds(g0, g1, z4, z4, 0);
#endif
}

// ---------------------------------------------------------------------------
// f32 -> f16 weight conversion
// ---------------------------------------------------------------------------
__global__ __launch_bounds__(256) void k_cvt(const float* __restrict__ in,
                                             _Float16* __restrict__ out, int n) {
    int i = (blockIdx.x * 256 + threadIdx.x) * 4;
    if (i < n) {
        float4 f = *(const float4*)(in + i);
        out[i + 0] = (_Float16)f.x;
        out[i + 1] = (_Float16)f.y;
        out[i + 2] = (_Float16)f.z;
        out[i + 3] = (_Float16)f.w;
    }
}

// ---------------------------------------------------------------------------
// LayerNorm over D=1024, one row per workgroup (256 thr = 8 waves), f16 out
// ---------------------------------------------------------------------------
__global__ __launch_bounds__(256) void k_layernorm(const float* __restrict__ x,
                                                   const float* __restrict__ g,
                                                   const float* __restrict__ b,
                                                   _Float16* __restrict__ out) {
    const int row  = blockIdx.x;
    const int tid  = threadIdx.x;
    const int lane = tid & 31, wave = tid >> 5;
    __shared__ float red[8];

    float4 f = ((const float4*)(x + (size_t)row * DM))[tid];
    float s = f.x + f.y + f.z + f.w;
#pragma unroll
    for (int o = 16; o > 0; o >>= 1) s += __shfl_xor(s, o, 32);
    if (lane == 0) red[wave] = s;
    __syncthreads();
    float mu = 0.f;
#pragma unroll
    for (int w = 0; w < 8; ++w) mu += red[w];
    mu *= (1.0f / DM);
    __syncthreads();

    float4 d;
    d.x = f.x - mu; d.y = f.y - mu; d.z = f.z - mu; d.w = f.w - mu;
    float s2 = d.x * d.x + d.y * d.y + d.z * d.z + d.w * d.w;
#pragma unroll
    for (int o = 16; o > 0; o >>= 1) s2 += __shfl_xor(s2, o, 32);
    if (lane == 0) red[wave] = s2;
    __syncthreads();
    float var = 0.f;
#pragma unroll
    for (int w = 0; w < 8; ++w) var += red[w];
    var *= (1.0f / DM);
    const float rs = rsqrtf(var + 1e-5f);

    const int c = tid * 4;
    _Float16* o4 = out + (size_t)row * DM + c;
    o4[0] = (_Float16)(d.x * rs * g[c + 0] + b[c + 0]);
    o4[1] = (_Float16)(d.y * rs * g[c + 1] + b[c + 1]);
    o4[2] = (_Float16)(d.z * rs * g[c + 2] + b[c + 2]);
    o4[3] = (_Float16)(d.w * rs * g[c + 3] + b[c + 3]);
}

// ---------------------------------------------------------------------------
// Tiled WMMA GEMM:  C[M,N] = act(A[M,K] * W[N,K]^T + bias) (+ resid)
// BM=BN=128, BK=32.  8 waves as 4(M)x2(N); wave = 32x64 = 2x4 wmma tiles.
// Tiles staged to LDS by the Tensor Data Mover (wave 0 issues, TENSORcnt).
// ACT: 0 none, 1 elu+1, 2 exact gelu.  OUT16: f16 vs f32 out.  RESID: +f32.
// ---------------------------------------------------------------------------
#define G_LDT 40   // padded LDS stride in f16 (80B) == TDM pad layout

template <int ACT, int OUT16, int RESID>
__global__ __launch_bounds__(256) void k_gemm(const _Float16* __restrict__ A,
                                              const _Float16* __restrict__ W,
                                              const float* __restrict__ bias,
                                              const float* __restrict__ resid,
                                              void* __restrict__ outp,
                                              int M, int N, int K) {
    __shared__ __align__(128) _Float16 sA[2][128 * G_LDT];
    __shared__ __align__(128) _Float16 sB[2][128 * G_LDT];

    const int tid  = threadIdx.x;
    const int lane = tid & 31, wave = tid >> 5;
    const int half = lane >> 4, l15 = lane & 15;
    const int wm = wave >> 1, wn = wave & 1;   // 4x2 wave grid
    const int m0 = blockIdx.y * 128, n0 = blockIdx.x * 128;

    const _Float16* gA = A + (size_t)m0 * K;   // tile base (row m0, k 0)
    const _Float16* gW = W + (size_t)n0 * K;
    const unsigned ldsA[2] = { (unsigned)(uintptr_t)&sA[0][0],
                               (unsigned)(uintptr_t)&sA[1][0] };
    const unsigned ldsB[2] = { (unsigned)(uintptr_t)&sB[0][0],
                               (unsigned)(uintptr_t)&sB[1][0] };
    const bool issuer = (tid < 32);            // wave 0 drives the TDM

    v8f acc[2][4] = {};

    if (issuer) {
        tdm_load_tile32(ldsA[0], gA, 128, (unsigned)K);
        tdm_load_tile32(ldsB[0], gW, 128, (unsigned)K);
    }

    const int nk = K >> 5;
    for (int kt = 0; kt < nk; ++kt) {
        const int buf = kt & 1;
        if (issuer) {
            if (kt + 1 < nk) {     // prefetch next tile into other buffer
                tdm_load_tile32(ldsA[buf ^ 1], gA + (size_t)(kt + 1) * 32,
                                128, (unsigned)K);
                tdm_load_tile32(ldsB[buf ^ 1], gW + (size_t)(kt + 1) * 32,
                                128, (unsigned)K);
                // TDM ops complete in order: <=2 outstanding => tile kt done
                __builtin_amdgcn_s_wait_tensorcnt(2);
            } else {
                __builtin_amdgcn_s_wait_tensorcnt(0);
            }
        }
        __syncthreads();   // tile kt visible to all waves

        v16h af[2], bf[4];
#pragma unroll
        for (int i = 0; i < 2; ++i)
            af[i] = ld_frag(sA[buf], wm * 32 + i * 16 + l15, 0, G_LDT, half);
#pragma unroll
        for (int j = 0; j < 4; ++j)
            bf[j] = ld_frag(sB[buf], wn * 64 + j * 16 + l15, 0, G_LDT, half);
#pragma unroll
        for (int i = 0; i < 2; ++i)
#pragma unroll
            for (int j = 0; j < 4; ++j)
                acc[i][j] = wmma16(af[i], bf[j], acc[i][j]);

        __syncthreads();   // all reads of buf done before TDM refills it
    }

    // epilogue: bias + activation (+ residual), scatter per C-layout
#pragma unroll
    for (int i = 0; i < 2; ++i) {
#pragma unroll
        for (int j = 0; j < 4; ++j) {
            const int n  = n0 + wn * 64 + j * 16 + l15;
            const float bv = bias[n];
#pragma unroll
            for (int e = 0; e < 8; ++e) {
                const int m = m0 + wm * 32 + i * 16 + half * 8 + e;
                float vv = acc[i][j][e] + bv;
                if (ACT == 1) vv = (vv > 0.f) ? (vv + 1.f) : __expf(vv);
                if (ACT == 2) vv = 0.5f * vv * (1.f + erff(vv * 0.70710678118f));
                if (RESID) vv += resid[(size_t)m * N + n];
                if (OUT16) ((_Float16*)outp)[(size_t)m * N + n] = (_Float16)vv;
                else       ((float*)outp)[(size_t)m * N + n]    = vv;
            }
        }
    }
}

// ---------------------------------------------------------------------------
// Chunked causal linear attention, one head per workgroup (grid=16, 256 thr).
// Chunk T=64:  A = QK^T (masked);  num = A_mask*V + Q*S_prev;  S += K^T*V;
// den_t = q_t.z_prev + rowsum(A_mask).  64x64 state in per-wave f32 accums.
// ---------------------------------------------------------------------------
#define A_LDT 72   // padded LDS stride (144B): conflict-free b128 rows

__global__ __launch_bounds__(256) void k_attn(const _Float16* __restrict__ q,
                                              const _Float16* __restrict__ k,
                                              const _Float16* __restrict__ v,
                                              _Float16* __restrict__ out) {
    __shared__ __align__(128) _Float16 sQ [64 * A_LDT];   // [t][d]
    __shared__ __align__(128) _Float16 sK [64 * A_LDT];   // [i][d]
    __shared__ __align__(128) _Float16 sKT[64 * A_LDT];   // [d][i]
    __shared__ __align__(128) _Float16 sVT[64 * A_LDT];   // [e][i]
    __shared__ __align__(128) _Float16 sAm[64 * A_LDT];   // masked A [t][i]
    __shared__ __align__(128) _Float16 sST[64 * A_LDT];   // state^T  [e][d]
    __shared__ float sZ[64];
    __shared__ float sDen[64];

    const int h    = blockIdx.x;
    const int tid  = threadIdx.x;
    const int lane = tid & 31, wave = tid >> 5;
    const int half = lane >> 4, l15 = lane & 15;
    const int tm   = wave >> 1;          // M-tile 0..3
    const int tnb  = (wave & 1) * 2;     // N-tile base {0,2}

    for (int i = tid; i < 64 * A_LDT; i += 256) sST[i] = (_Float16)0.f;
    if (tid < 64) sZ[tid] = 0.f;
    v8f accS[2] = {};
    __syncthreads();

    const int lr = tid >> 2;             // loader row 0..63
    const int lc = (tid & 3) * 16;       // loader col {0,16,32,48}

    for (int c = 0; c < SEQ / 64; ++c) {
        const int t0 = c * 64;
        // ---- stage Q,K,K^T,V^T into LDS ----
        {
            const size_t gb = (size_t)(t0 + lr) * DM + h * HD + lc;
            v8h q0 = *(const v8h*)(q + gb), q1 = *(const v8h*)(q + gb + 8);
            v8h k0v = *(const v8h*)(k + gb), k1v = *(const v8h*)(k + gb + 8);
            v8h v0 = *(const v8h*)(v + gb), v1 = *(const v8h*)(v + gb + 8);
            if (c + 1 < SEQ / 64) {              // gfx1250 global_prefetch_b8
                __builtin_prefetch(q + gb + (size_t)64 * DM, 0, 1);
                __builtin_prefetch(k + gb + (size_t)64 * DM, 0, 1);
                __builtin_prefetch(v + gb + (size_t)64 * DM, 0, 1);
            }
            _Float16* dq = &sQ[lr * A_LDT + lc];
            *(v8h*)dq = q0; *(v8h*)(dq + 8) = q1;
            _Float16* dk = &sK[lr * A_LDT + lc];
            *(v8h*)dk = k0v; *(v8h*)(dk + 8) = k1v;
#pragma unroll
            for (int j = 0; j < 8; ++j) {
                sKT[(lc + j)     * A_LDT + lr] = k0v[j];
                sKT[(lc + 8 + j) * A_LDT + lr] = k1v[j];
                sVT[(lc + j)     * A_LDT + lr] = v0[j];
                sVT[(lc + 8 + j) * A_LDT + lr] = v1[j];
            }
        }
        __syncthreads();                                   // B1

        // ---- G1: A = Q K^T (contract over d) ----
        v8f accA[2] = {};
#pragma unroll
        for (int kk = 0; kk < 64; kk += 32) {
            v16h af = ld_frag(sQ, tm * 16 + l15, kk, A_LDT, half);
#pragma unroll
            for (int jj = 0; jj < 2; ++jj) {
                v16h bf = ld_frag(sK, (tnb + jj) * 16 + l15, kk, A_LDT, half);
                accA[jj] = wmma16(af, bf, accA[jj]);
            }
        }
        // causal mask (keep i <= t), spill masked A to LDS as f16
#pragma unroll
        for (int jj = 0; jj < 2; ++jj) {
            const int i_loc = (tnb + jj) * 16 + l15;
#pragma unroll
            for (int e = 0; e < 8; ++e) {
                const int t_loc = tm * 16 + half * 8 + e;
                const float vv = (i_loc <= t_loc) ? accA[jj][e] : 0.f;
                sAm[t_loc * A_LDT + i_loc] = (_Float16)vv;
            }
        }
        __syncthreads();                                   // B2

        // ---- denominator: q.z_prev + rowsum(A_mask) + eps ----
        if (tid < 64) {
            float den = 1e-6f;
#pragma unroll 8
            for (int d = 0; d < 64; ++d)
                den += (float)sQ[tid * A_LDT + d] * sZ[d];
#pragma unroll 8
            for (int i = 0; i < 64; ++i)
                den += (float)sAm[tid * A_LDT + i];
            sDen[tid] = den;
        }

        // ---- G2 (intra) + G3 (inter) into num;  G4 state update ----
        v8f accN[2] = {};
#pragma unroll
        for (int kk = 0; kk < 64; kk += 32) {
            v16h aQ = ld_frag(sQ,  tm * 16 + l15, kk, A_LDT, half);
            v16h aA = ld_frag(sAm, tm * 16 + l15, kk, A_LDT, half);
            v16h aK = ld_frag(sKT, tm * 16 + l15, kk, A_LDT, half);
#pragma unroll
            for (int jj = 0; jj < 2; ++jj) {
                v16h bV = ld_frag(sVT, (tnb + jj) * 16 + l15, kk, A_LDT, half);
                v16h bS = ld_frag(sST, (tnb + jj) * 16 + l15, kk, A_LDT, half);
                accN[jj] = wmma16(aA, bV, accN[jj]);       // A_mask * V
                accN[jj] = wmma16(aQ, bS, accN[jj]);       // Q * S_prev
                accS[jj] = wmma16(aK, bV, accS[jj]);       // S += K^T V
            }
        }
        __syncthreads();                                   // B3

        // ---- publish new state^T + z update ----
#pragma unroll
        for (int jj = 0; jj < 2; ++jj) {
            const int eo = (tnb + jj) * 16 + l15;
#pragma unroll
            for (int e = 0; e < 8; ++e) {
                const int dd = tm * 16 + half * 8 + e;
                sST[eo * A_LDT + dd] = (_Float16)accS[jj][e];
            }
        }
        if (tid < 64) {
            float zt = sZ[tid];
#pragma unroll 8
            for (int i = 0; i < 64; ++i) zt += (float)sKT[tid * A_LDT + i];
            sZ[tid] = zt;
        }

        // ---- output: num / den ----
#pragma unroll
        for (int jj = 0; jj < 2; ++jj) {
            const int eo = (tnb + jj) * 16 + l15;
#pragma unroll
            for (int e = 0; e < 8; ++e) {
                const int t_loc = tm * 16 + half * 8 + e;
                const float oo = accN[jj][e] / sDen[t_loc];
                out[(size_t)(t0 + t_loc) * DM + h * HD + eo] = (_Float16)oo;
            }
        }
        __syncthreads();                                   // B4
    }
}

// ---------------------------------------------------------------------------
extern "C" void kernel_launch(void* const* d_in, const int* in_sizes, int n_in,
                              void* d_out, int out_size, void* d_ws, size_t ws_size,
                              hipStream_t stream) {
    (void)in_sizes; (void)n_in; (void)out_size; (void)ws_size;
    const float* x   = (const float*)d_in[0];
    const float* Wq  = (const float*)d_in[1];
    const float* bq  = (const float*)d_in[2];
    const float* Wk  = (const float*)d_in[3];
    const float* bk  = (const float*)d_in[4];
    const float* Wv  = (const float*)d_in[5];
    const float* bv  = (const float*)d_in[6];
    const float* Wo  = (const float*)d_in[7];
    const float* bo  = (const float*)d_in[8];
    const float* W1  = (const float*)d_in[9];
    const float* bb1 = (const float*)d_in[10];
    const float* W2  = (const float*)d_in[11];
    const float* bb2 = (const float*)d_in[12];
    const float* g1  = (const float*)d_in[13];
    const float* be1 = (const float*)d_in[14];
    const float* g2  = (const float*)d_in[15];
    const float* be2 = (const float*)d_in[16];
    float* out = (float*)d_out;

    char* ws = (char*)d_ws;
    size_t off = 0;
    auto alloc = [&](size_t bytes) {
        void* p = ws + off;
        off += (bytes + 255) & ~(size_t)255;
        return p;
    };
    _Float16* wq16 = (_Float16*)alloc((size_t)DM * DM * 2);
    _Float16* wk16 = (_Float16*)alloc((size_t)DM * DM * 2);
    _Float16* wv16 = (_Float16*)alloc((size_t)DM * DM * 2);
    _Float16* wo16 = (_Float16*)alloc((size_t)DM * DM * 2);
    _Float16* w116 = (_Float16*)alloc((size_t)FFD * DM * 2);
    _Float16* w216 = (_Float16*)alloc((size_t)DM * FFD * 2);
    _Float16* h1   = (_Float16*)alloc((size_t)SEQ * DM * 2);   // reused as attn
    _Float16* qb   = (_Float16*)alloc((size_t)SEQ * DM * 2);   // reused as h2
    _Float16* kb   = (_Float16*)alloc((size_t)SEQ * DM * 2);
    _Float16* vb   = (_Float16*)alloc((size_t)SEQ * DM * 2);
    _Float16* ffb  = (_Float16*)alloc((size_t)SEQ * FFD * 2);
    _Float16* attnb = h1;   // h1 dead after V projection
    _Float16* h2    = qb;   // q dead after attention

    // weights -> f16
    k_cvt<<<(DM * DM) / 1024, 256, 0, stream>>>(Wq, wq16, DM * DM);
    k_cvt<<<(DM * DM) / 1024, 256, 0, stream>>>(Wk, wk16, DM * DM);
    k_cvt<<<(DM * DM) / 1024, 256, 0, stream>>>(Wv, wv16, DM * DM);
    k_cvt<<<(DM * DM) / 1024, 256, 0, stream>>>(Wo, wo16, DM * DM);
    k_cvt<<<(FFD * DM) / 1024, 256, 0, stream>>>(W1, w116, FFD * DM);
    k_cvt<<<(DM * FFD) / 1024, 256, 0, stream>>>(W2, w216, DM * FFD);

    // LN1
    k_layernorm<<<SEQ, 256, 0, stream>>>(x, g1, be1, h1);

    // Q/K/V projections (elu+1 fused into Q,K)
    dim3 gP(DM / 128, SEQ / 128);
    k_gemm<1, 1, 0><<<gP, 256, 0, stream>>>(h1, wq16, bq, nullptr, qb, SEQ, DM, DM);
    k_gemm<1, 1, 0><<<gP, 256, 0, stream>>>(h1, wk16, bk, nullptr, kb, SEQ, DM, DM);
    k_gemm<0, 1, 0><<<gP, 256, 0, stream>>>(h1, wv16, bv, nullptr, vb, SEQ, DM, DM);

    // chunked causal linear attention
    k_attn<<<NH, 256, 0, stream>>>(qb, kb, vb, attnb);

    // output projection + residual (x) -> d_out (= x2)
    k_gemm<0, 0, 1><<<gP, 256, 0, stream>>>(attnb, wo16, bo, x, out, SEQ, DM, DM);

    // LN2
    k_layernorm<<<SEQ, 256, 0, stream>>>(out, g2, be2, h2);

    // FFN
    dim3 gF1(FFD / 128, SEQ / 128);
    k_gemm<2, 1, 0><<<gF1, 256, 0, stream>>>(h2, w116, bb1, nullptr, ffb, SEQ, FFD, DM);
    dim3 gF2(DM / 128, SEQ / 128);
    k_gemm<0, 0, 1><<<gF2, 256, 0, stream>>>(ffb, w216, bb2, out, out, SEQ, DM, FFD);
}
